// CoLAEncoderLayer_28432683499590
// MI455X (gfx1250) — compile-verified, matrix operand order checked by reading, not run on hardware
//
#include <hip/hip_runtime.h>
#include <hip/hip_bf16.h>
#include <math.h>

// ---------------- types ----------------
typedef __attribute__((ext_vector_type(16))) _Float16 v16h;
typedef __attribute__((ext_vector_type(8)))  _Float16 v8h;
typedef __attribute__((ext_vector_type(4)))  _Float16 v4h;
typedef __attribute__((ext_vector_type(8)))  float    v8f;
typedef __attribute__((ext_vector_type(4)))  float    v4f;

#define DD     1024
#define ALPHA  64
#define NROW   32768          // B*L = 8*4096
#define SCALE  0.03125f       // 1/sqrt(1024)
#define LNEPS  1e-6f

// ---------------- helpers ----------------
__device__ __forceinline__ v8f wmma_f16(v16h a, v16h b, v8f c) {
  return __builtin_amdgcn_wmma_f32_16x16x32_f16(false, a, false, b, (short)0, c,
                                                false, false);
}

__device__ __forceinline__ v8f zero8() {
  v8f z;
#pragma unroll
  for (int i = 0; i < 8; ++i) z[i] = 0.0f;
  return z;
}

__device__ __forceinline__ v16h combine8(v8h lo, v8h hi) {
  v16h r;
#pragma unroll
  for (int j = 0; j < 8; ++j) { r[j] = lo[j]; r[j + 8] = hi[j]; }
  return r;
}

// A fragment (16x32 f16) from a row-major f16 row pointer, k-window at kbase.
// ISA A layout: lane half hs covers K in [hs*8, hs*8+7] U [16+hs*8, 16+hs*8+7].
__device__ __forceinline__ v16h load_a_f16(const _Float16* row, int kbase, int hs) {
  v8h lo = *(const v8h*)(row + kbase + hs * 8);
  v8h hi = *(const v8h*)(row + kbase + 16 + hs * 8);
  return combine8(lo, hi);
}

// Same A fragment, but converting from an fp32 row on the fly.
__device__ __forceinline__ v16h load_a_f32(const float* row, int kbase, int hs) {
  v4f a0 = *(const v4f*)(row + kbase + hs * 8);
  v4f a1 = *(const v4f*)(row + kbase + hs * 8 + 4);
  v4f b0 = *(const v4f*)(row + kbase + 16 + hs * 8);
  v4f b1 = *(const v4f*)(row + kbase + 16 + hs * 8 + 4);
  v16h r;
#pragma unroll
  for (int j = 0; j < 4; ++j) {
    r[j]      = (_Float16)a0[j];
    r[j + 4]  = (_Float16)a1[j];
    r[j + 8]  = (_Float16)b0[j];
    r[j + 12] = (_Float16)b1[j];
  }
  return r;
}

// Swizzled B-layout offset for element (kk, n) of a KxN f16 matrix, NT = N/16.
// Fragment read side: lane reads 16 contiguous halves at ((kb*NT+nt)*512 + lane*16).
__device__ __forceinline__ size_t bswz(int kk, int n, int NT) {
  int kb = kk >> 5, k = kk & 31;
  int hs = (k >> 3) & 1;
  int i  = (k & 7) + ((k & 16) >> 1);
  int lane = (n & 15) + (hs << 4);
  int nt = n >> 4;
  return ((size_t)(kb * NT + nt) * 512 + (size_t)lane * 16 + i);
}

// Async global->LDS staging: each thread copies 256 contiguous bytes with 16
// GLOBAL_LOAD_ASYNC_TO_LDS_B128 issues (GVS mode: SGPR64 base + per-lane u32 off).
__device__ __forceinline__ void stage_async_256(const void* gbase, unsigned gloff,
                                                unsigned ldsoff) {
#pragma unroll
  for (int j = 0; j < 16; ++j) {
    asm volatile("global_load_async_to_lds_b128 %0, %1, %2"
                 :
                 : "v"(ldsoff + j * 16), "v"(gloff + j * 16), "s"(gbase)
                 : "memory");
  }
}

__device__ __forceinline__ void wait_async0() {
  asm volatile("s_wait_asynccnt 0" ::: "memory");
}

// ---------------- precompute kernels ----------------

// W2[d][a] = sum_e W_Q[e,d] * C_K[e,a]  -> swizzled f16 (K=1024, N=64)
__global__ __launch_bounds__(256) void k_w2(const float* __restrict__ WQ,
                                            const float* __restrict__ CK,
                                            _Float16* __restrict__ W2sw) {
  int t = blockIdx.x * 256 + threadIdx.x;   // 0..65535
  int d = t >> 6, a = t & 63;
  float s = 0.0f;
  for (int e = 0; e < DD; ++e)
    s = fmaf(WQ[(size_t)e * DD + d], CK[(size_t)e * ALPHA + a], s);
  W2sw[bswz(d, a, 4)] = (_Float16)s;
}

// b2[a] = sum_e b_Q[e] * C_K[e,a]
__global__ void k_b2(const float* __restrict__ bQ, const float* __restrict__ CK,
                     float* __restrict__ b2) {
  int a = threadIdx.x;   // 64 threads
  float s = 0.0f;
  for (int e = 0; e < DD; ++e) s = fmaf(bQ[e], CK[(size_t)e * ALPHA + a], s);
  b2[a] = s;
}

// CVsw: B[k=a][n=d] = C_V[d,a] swizzled f16 (K=64, N=1024)
__global__ __launch_bounds__(256) void k_cvsw(const float* __restrict__ CV,
                                              _Float16* __restrict__ CVsw) {
  int t = blockIdx.x * 256 + threadIdx.x;   // 0..65535
  int a = t >> 10, dcol = t & 1023;
  CVsw[bswz(a, dcol, 64)] = (_Float16)CV[(size_t)dcol * ALPHA + a];
}

// Wcsw: B[k=d][n=e] = Wc[e,d] swizzled f16 (K=1024, N=1024)
__global__ __launch_bounds__(256) void k_wcsw(const float* __restrict__ Wc,
                                              _Float16* __restrict__ Wcsw) {
  int t = blockIdx.x * 256 + threadIdx.x;   // 0..1048575
  int d = t >> 10, e = t & 1023;
  Wcsw[bswz(d, e, 64)] = (_Float16)Wc[(size_t)e * DD + d];
}

// ------- fused: scores GEMM -> softmax -> attn GEMM -> residual -> LN1 -> h16 -------
// 4 waves * 16 rows = 64 rows per workgroup. LDS pool ~145KB (< 320KB WGP LDS).
#define AT_ABUF   0                       // _Float16 [4][16][64]   (8192 B)
#define AT_RED1   8192                    // float    [4][16][16]   (4096 B)
#define AT_RED2   12288                   // float    [4][16][16]   (4096 B)
#define AT_STATS  16384                   // float    [4][16][2]    (512 B)
#define AT_HTILE  16896                   // _Float16 [4][16][1024] (131072 B)
#define AT_SBUF   16896                   // float [4][16][64] ALIASES Htile (time-disjoint)
#define AT_POOL   (16896 + 131072)

__global__ __launch_bounds__(128) void k_attn(
    const float* __restrict__ x, const int* __restrict__ mask,
    const _Float16* __restrict__ W2sw, const float* __restrict__ b2,
    const _Float16* __restrict__ CVsw, const float* __restrict__ g1,
    const float* __restrict__ be1, _Float16* __restrict__ h16) {
  __shared__ __align__(16) char pool[AT_POOL];
  _Float16* Abuf  = (_Float16*)(pool + AT_ABUF);
  float*    Red1  = (float*)(pool + AT_RED1);
  float*    Red2  = (float*)(pool + AT_RED2);
  float*    Stat  = (float*)(pool + AT_STATS);
  _Float16* Htile = (_Float16*)(pool + AT_HTILE);
  float*    Sbuf  = (float*)(pool + AT_SBUF);

  const int tid = threadIdx.x;
  const int wave = tid >> 5, lane = tid & 31;
  const int m = lane & 15, hs = lane >> 4;
  const int rowmod = (lane < 16) ? 0 : 8;
  const int r0 = blockIdx.x * 64 + wave * 16;

  // ---- phase 1: scores(16x64) = X(16x1024) @ W2, A converted f32->f16 on the fly ----
  const float* xrow = x + (size_t)(r0 + m) * DD;
  v8f accS[4];
#pragma unroll
  for (int nt = 0; nt < 4; ++nt) accS[nt] = zero8();
  for (int kb = 0; kb < 32; ++kb) {
    if (kb < 31) {
      __builtin_prefetch(xrow + (kb + 1) * 32 + hs * 8, 0, 3);
      __builtin_prefetch(W2sw + ((size_t)((kb + 1) * 4) * 512 + lane * 16), 0, 3);
    }
    v16h af = load_a_f32(xrow, kb * 32, hs);
#pragma unroll
    for (int nt = 0; nt < 4; ++nt) {
      v16h bf = *(const v16h*)(W2sw + ((size_t)(kb * 4 + nt) * 512 + lane * 16));
      accS[nt] = wmma_f16(af, bf, accS[nt]);
    }
  }

  // ---- phase 2: scatter C layout to LDS, per-row softmax ----
#pragma unroll
  for (int r = 0; r < 8; ++r)
#pragma unroll
    for (int nt = 0; nt < 4; ++nt)
      Sbuf[(wave * 16 + r + rowmod) * 64 + nt * 16 + m] = accS[nt][r];
  __syncthreads();

  if (lane < 16) {
    const float mk = (float)mask[r0 + m];
    float sv[64];
    float mx = -1e30f;
#pragma unroll
    for (int a = 0; a < 64; ++a) {
      float v = Sbuf[(wave * 16 + m) * 64 + a] * SCALE + b2[a];
      sv[a] = v;
      mx = fmaxf(mx, v);
    }
    float sum = 0.0f;
#pragma unroll
    for (int a = 0; a < 64; ++a) {
      float e = __expf(sv[a] - mx);
      sv[a] = e;
      sum += e;
    }
    const float f = mk / sum;   // mask applied post-softmax (matches ref)
#pragma unroll
    for (int a = 0; a < 64; ++a)
      Abuf[(wave * 16 + m) * 64 + a] = (_Float16)(sv[a] * f);
  }
  __syncthreads();

  // ---- phase 3: attn(16x1024) = A(16x64) @ C_V^T ; h = x + attn -> LDS; stats ----
  float p1[8], p2[8];
#pragma unroll
  for (int r = 0; r < 8; ++r) { p1[r] = 0.0f; p2[r] = 0.0f; }
  const _Float16* arow = Abuf + (wave * 16 + m) * 64;
  for (int nc = 0; nc < 4; ++nc) {
#pragma unroll
    for (int nt = 0; nt < 16; ++nt) {
      v8f acc = zero8();
      const int ntg = nc * 16 + nt;
#pragma unroll
      for (int kb = 0; kb < 2; ++kb) {
        v16h af = load_a_f16(arow, kb * 32, hs);
        v16h bf = *(const v16h*)(CVsw + ((size_t)(kb * 64 + ntg) * 512 + lane * 16));
        acc = wmma_f16(af, bf, acc);
      }
      const int col = ntg * 16 + m;
#pragma unroll
      for (int r = 0; r < 8; ++r) {
        size_t idx = (size_t)(r0 + r + rowmod) * DD + col;
        float h = x[idx] + acc[r];
        Htile[(wave * 16 + r + rowmod) * 1024 + col] = (_Float16)h;
        p1[r] += h;
        p2[r] += h * h;
      }
    }
  }

  // ---- cross-lane row-stat reduction -> per-row (mean, 1/(std+eps)) in LDS ----
#pragma unroll
  for (int r = 0; r < 8; ++r) {
    Red1[(wave * 16 + r + rowmod) * 16 + m] = p1[r];
    Red2[(wave * 16 + r + rowmod) * 16 + m] = p2[r];
  }
  __syncthreads();
  if (lane < 16) {
    float s1 = 0.0f, s2 = 0.0f;
#pragma unroll
    for (int j = 0; j < 16; ++j) {
      s1 += Red1[(wave * 16 + m) * 16 + j];
      s2 += Red2[(wave * 16 + m) * 16 + j];
    }
    float mean = s1 * (1.0f / 1024.0f);
    float var  = (s2 - s1 * s1 * (1.0f / 1024.0f)) * (1.0f / 1023.0f);
    Stat[(wave * 16 + m) * 2]     = mean;
    Stat[(wave * 16 + m) * 2 + 1] = 1.0f / (sqrtf(var) + LNEPS);
  }
  __syncthreads();

  // ---- phase 4: apply LN1, write h16 (coalesced, 4 halves/lane) ----
  for (int it = 0; it < 128; ++it) {
    int flat = it * 128 + lane * 4;          // wave-local element index
    int row = flat >> 10, col = flat & 1023;
    float mean = Stat[(wave * 16 + row) * 2];
    float inv  = Stat[(wave * 16 + row) * 2 + 1];
    v4h ht = *(const v4h*)(Htile + (wave * 16 + row) * 1024 + col);
    v4h r;
#pragma unroll
    for (int j = 0; j < 4; ++j)
      r[j] = (_Float16)(g1[col + j] * ((float)ht[j] - mean) * inv + be1[col + j]);
    *(v4h*)(h16 + (size_t)(r0 + row) * DD + col) = r;
  }
}

// ------- fused: conv GEMM (async-LDS-staged B) + bias + LeakyReLU + residual + LN2 -------
// 8 waves: 2 row-groups x 4 col-quarters -> 32 rows x 1024 cols per workgroup.
// B k-slice (32x1024 f16 = 64KB) is DMA'd into LDS with GLOBAL_LOAD_ASYNC_TO_LDS_B128,
// double buffered; the epilogue Ttile ALIASES the B buffers (time-disjoint).
#define CV_BBUF   0                       // _Float16 [2][32768]  (131072 B)
#define CV_TTILE  0                       // float [32][1024] ALIASES B buffers
#define CV_RED1   131072                  // float [2][16][64] (8192 B)
#define CV_RED2   139264                  // float [2][16][64] (8192 B)
#define CV_STATS  147456                  // float [32][2]     (256 B)
#define CV_POOL   (147456 + 256)

__global__ __launch_bounds__(256) void k_conv(
    const _Float16* __restrict__ h16, const _Float16* __restrict__ Wcsw,
    const float* __restrict__ bc, const float* __restrict__ g2,
    const float* __restrict__ be2, float* __restrict__ out) {
  __shared__ __align__(16) char pool[CV_POOL];
  const _Float16* Bs = (const _Float16*)(pool + CV_BBUF);
  float* Ttile = (float*)(pool + CV_TTILE);
  float* Red1  = (float*)(pool + CV_RED1);
  float* Red2  = (float*)(pool + CV_RED2);
  float* Stat  = (float*)(pool + CV_STATS);

  const int tid = threadIdx.x;
  const int wave = tid >> 5, lane = tid & 31;
  const int m = lane & 15, hs = lane >> 4;
  const int rowmod = (lane < 16) ? 0 : 8;
  const int mg = wave & 1, nq = wave >> 1;
  const int r0 = blockIdx.x * 32 + mg * 16;

  const unsigned ldsB = (unsigned)(uintptr_t)(pool + CV_BBUF);  // LDS offset = addr[31:0]
  const unsigned toff = (unsigned)tid * 256u;

  const _Float16* hrow = h16 + (size_t)(r0 + m) * DD;
  v8f acc[16];
#pragma unroll
  for (int nt = 0; nt < 16; ++nt) acc[nt] = zero8();

  // prologue: DMA k-slice 0 into buffer 0
  stage_async_256(Wcsw, toff, ldsB + toff);
  wait_async0();
  __syncthreads();

  for (int kb = 0; kb < 32; ++kb) {
    // DMA next k-slice into the other buffer while computing this one
    if (kb < 31)
      stage_async_256(Wcsw, (unsigned)((kb + 1) * 65536) + toff,
                      ldsB + (unsigned)(((kb + 1) & 1) * 65536) + toff);
    if (kb < 31) __builtin_prefetch(hrow + (kb + 1) * 32 + hs * 8, 0, 3);

    const _Float16* bslice = Bs + (size_t)(kb & 1) * 32768;
    v16h af = load_a_f16(hrow, kb * 32, hs);
#pragma unroll
    for (int nt = 0; nt < 16; ++nt) {
      v16h bf = *(const v16h*)(bslice + (size_t)(nq * 16 + nt) * 512 + lane * 16);
      acc[nt] = wmma_f16(af, bf, acc[nt]);
    }

    wait_async0();     // my async issues for kb+1 are complete
    __syncthreads();   // everyone done reading slice kb and staging kb+1
  }

  // epilogue: bias, LeakyReLU, residual (h16 is post-LN1 h), t -> LDS (aliases B), stats
  float p1[8], p2[8];
#pragma unroll
  for (int r = 0; r < 8; ++r) { p1[r] = 0.0f; p2[r] = 0.0f; }
#pragma unroll
  for (int nt = 0; nt < 16; ++nt) {
    const int col = nq * 256 + nt * 16 + m;
    const float bcv = bc[col];
#pragma unroll
    for (int r = 0; r < 8; ++r) {
      const int rloc = mg * 16 + r + rowmod;
      size_t idx = (size_t)(r0 + r + rowmod) * DD + col;
      float conv = acc[nt][r] + bcv;
      float act = (conv > 0.0f) ? conv : 0.01f * conv;
      float t = (float)h16[idx] + act;
      Ttile[rloc * 1024 + col] = t;
      p1[r] += t;
      p2[r] += t * t;
    }
  }

#pragma unroll
  for (int r = 0; r < 8; ++r) {
    Red1[(mg * 16 + r + rowmod) * 64 + nq * 16 + m] = p1[r];
    Red2[(mg * 16 + r + rowmod) * 64 + nq * 16 + m] = p2[r];
  }
  __syncthreads();
  if (tid < 32) {
    float s1 = 0.0f, s2 = 0.0f;
    for (int j = 0; j < 64; ++j) {
      s1 += Red1[tid * 64 + j];
      s2 += Red2[tid * 64 + j];
    }
    float mean = s1 * (1.0f / 1024.0f);
    float var  = (s2 - s1 * s1 * (1.0f / 1024.0f)) * (1.0f / 1023.0f);
    Stat[tid * 2]     = mean;
    Stat[tid * 2 + 1] = 1.0f / (sqrtf(var) + LNEPS);
  }
  __syncthreads();

  // apply LN2, write out (one full row per iteration: 256 threads * 4 floats)
  for (int it = 0; it < 32; ++it) {
    const int col = tid * 4;
    float mean = Stat[it * 2], inv = Stat[it * 2 + 1];
    const float* trow = Ttile + it * 1024 + col;
    v4f o;
#pragma unroll
    for (int j = 0; j < 4; ++j)
      o[j] = g2[col + j] * (trow[j] - mean) * inv + be2[col + j];
    *(v4f*)(out + (size_t)(blockIdx.x * 32 + it) * DD + col) = o;
  }
}

// ---------------- launch ----------------
extern "C" void kernel_launch(void* const* d_in, const int* in_sizes, int n_in,
                              void* d_out, int out_size, void* d_ws, size_t ws_size,
                              hipStream_t stream) {
  const float* x    = (const float*)d_in[0];
  const int*   mask = (const int*)d_in[1];
  const float* WQ   = (const float*)d_in[2];
  const float* bQ   = (const float*)d_in[3];
  const float* CK   = (const float*)d_in[4];
  const float* CV   = (const float*)d_in[5];
  const float* g1   = (const float*)d_in[6];
  const float* be1  = (const float*)d_in[7];
  const float* Wc   = (const float*)d_in[8];
  const float* bc   = (const float*)d_in[9];
  const float* g2   = (const float*)d_in[10];
  const float* be2  = (const float*)d_in[11];
  float* out = (float*)d_out;

  // workspace layout (256B-aligned offsets), ~69.5 MB total
  char* ws = (char*)d_ws;
  _Float16* W2sw = (_Float16*)(ws + 0);           // 131072 B
  float*    b2   = (float*)   (ws + 131072);      // 256 B
  _Float16* CVsw = (_Float16*)(ws + 131328);      // 131072 B
  _Float16* Wcsw = (_Float16*)(ws + 262400);      // 2097152 B
  _Float16* h16  = (_Float16*)(ws + 2359552);     // 67108864 B

  // fold W_Q into W2 = W_Q^T @ C_K, plus f16 swizzled weight conversions
  k_w2  <<<256,  256, 0, stream>>>(WQ, CK, W2sw);
  k_b2  <<<1,    64,  0, stream>>>(bQ, CK, b2);
  k_cvsw<<<256,  256, 0, stream>>>(CV, CVsw);
  k_wcsw<<<4096, 256, 0, stream>>>(Wc, Wcsw);

  // fused attention: scores -> softmax -> attn -> residual -> LN1 -> h16 (f16)
  k_attn<<<512, 128, 0, stream>>>(x, mask, W2sw, b2, CVsw, g1, be1, h16);

  // fused FFN: async-staged conv GEMM -> bias -> LeakyReLU -> residual -> LN2 -> out
  k_conv<<<1024, 256, 0, stream>>>(h16, Wcsw, bc, g2, be2, out);
}
